// W8A8LinearLayer_14611478741504
// MI455X (gfx1250) — 8-bit matrix operand order repaired and checked by reading, compiled, not run on hardware
//
#include <hip/hip_runtime.h>
#include <hip/hip_bf16.h>

typedef __attribute__((ext_vector_type(8))) int v8i;
typedef __attribute__((ext_vector_type(4))) int v4i;

typedef __attribute__((address_space(1))) const signed char* gcptr;
typedef __attribute__((address_space(1))) const v4i*         gc_v4i;

__device__ __forceinline__ gcptr to_global(const signed char* p) {
    return (gcptr)(unsigned long long)p;
}
__device__ __forceinline__ v4i gload16(gcptr p) {
    return *(gc_v4i)p;
}


__device__ __forceinline__ size_t a_swizzle(int tileIdx, int r, int j) {
    const int lane = (r & 15) | ((j & 1) << 4);
    const int off  = ((j & 4) ? 512 : 0) + ((j & 2) ? 8 : 0);
    return (size_t)tileIdx * 1024 + lane * 16 + off;
}

__device__ __forceinline__ size_t b_swizzle(int tileIdx, int c, int j) {
    const int lane = (c & 15) | ((j & 1) << 4);
    const int off  = ((j & 4) ? 512 : 0) + ((j & 2) ? 8 : 0);
    return (size_t)tileIdx * 1024 + lane * 16 + off;
}

__global__ __launch_bounds__(256) void quantize_act(const float* __restrict__ x,
                                                    signed char* __restrict__ xq,
                                                    float* __restrict__ sact,
                                                    int K) {
    const int m   = blockIdx.x;
    const int tid = threadIdx.x;
    const int KT  = K >> 6;
    const float* row = x + (size_t)m * K;

    float tmax = 0.0f;
    const int k4 = K >> 2;
    for (int c = tid; c < k4; c += 256) {
        float4 f = ((const float4*)row)[c];
        tmax = fmaxf(tmax, fmaxf(fmaxf(fabsf(f.x), fabsf(f.y)),
                                 fmaxf(fabsf(f.z), fabsf(f.w))));
    }
    __shared__ float red[256];
    red[tid] = tmax;
    __syncthreads();
    #pragma unroll
    for (int s = 128; s > 0; s >>= 1) {
        if (tid < s) red[tid] = fmaxf(red[tid], red[tid + s]);
        __syncthreads();
    }
    const float mx  = fmaxf(red[0], 1e-8f);
    const float inv = 127.0f / mx;
    if (tid == 0) sact[m] = mx * (1.0f / 127.0f);

    const int k8 = K >> 3;
    const int tm = m >> 4;
    for (int c = tid; c < k8; c += 256) {
        float4 f0 = ((const float4*)row)[2 * c];
        float4 f1 = ((const float4*)row)[2 * c + 1];
        int q0 = (int)fminf(fmaxf(rintf(f0.x * inv), -128.0f), 127.0f);
        int q1 = (int)fminf(fmaxf(rintf(f0.y * inv), -128.0f), 127.0f);
        int q2 = (int)fminf(fmaxf(rintf(f0.z * inv), -128.0f), 127.0f);
        int q3 = (int)fminf(fmaxf(rintf(f0.w * inv), -128.0f), 127.0f);
        int q4 = (int)fminf(fmaxf(rintf(f1.x * inv), -128.0f), 127.0f);
        int q5 = (int)fminf(fmaxf(rintf(f1.y * inv), -128.0f), 127.0f);
        int q6 = (int)fminf(fmaxf(rintf(f1.z * inv), -128.0f), 127.0f);
        int q7 = (int)fminf(fmaxf(rintf(f1.w * inv), -128.0f), 127.0f);
        uint2 pk;
        pk.x = (q0 & 0xff) | ((q1 & 0xff) << 8) | ((q2 & 0xff) << 16) | ((q3 & 0xff) << 24);
        pk.y = (q4 & 0xff) | ((q5 & 0xff) << 8) | ((q6 & 0xff) << 16) | ((q7 & 0xff) << 24);
        const int tk = c >> 3;
        const int j  = c & 7;
        *(uint2*)(xq + a_swizzle(tm * KT + tk, m & 15, j)) = pk;
    }
}

__global__ __launch_bounds__(256) void pack_w(const int* __restrict__ wq,
                                              signed char* __restrict__ w8,
                                              int K) {
    const int n  = blockIdx.y;
    const int c  = blockIdx.x * 256 + threadIdx.x;
    const int KT = K >> 6;
    const int k  = c << 3;

    const int4* p = (const int4*)(wq + (size_t)n * K + k);
    int4 w0 = p[0];
    int4 w1 = p[1];
    uint2 pk;
    pk.x = (w0.x & 0xff) | ((w0.y & 0xff) << 8) | ((w0.z & 0xff) << 16) | ((w0.w & 0xff) << 24);
    pk.y = (w1.x & 0xff) | ((w1.y & 0xff) << 8) | ((w1.z & 0xff) << 16) | ((w1.w & 0xff) << 24);

    const int tn = n >> 4;
    const int tk = k >> 6;
    const int j  = (k & 63) >> 3;
    *(uint2*)(w8 + b_swizzle(tn * KT + tk, n & 15, j)) = pk;
}

__global__ __launch_bounds__(256) void w8a8_gemm(const signed char* __restrict__ Aq,
                                                 const signed char* __restrict__ Wq,
                                                 const float* __restrict__ sact,
                                                 const float* __restrict__ wscale,
                                                 const float* __restrict__ bias,
                                                 float* __restrict__ out,
                                                 int M, int N, int K) {
    const int KT   = K >> 6;
    const int tid  = threadIdx.x;
    const int wid  = tid >> 5;
    const int lane = tid & 31;
    const int l15  = lane & 15;
    const int lhi  = lane >> 4;

    const int bm = blockIdx.y * 128;
    const int bn = blockIdx.x * 256;
    const int wm = bm + (wid & 1) * 64;
    const int wn = bn + (wid >> 1) * 64;

    gcptr pA[4];
    gcptr pB[4];
    #pragma unroll
    for (int t = 0; t < 4; ++t) {
        pA[t] = to_global(Aq) + (size_t)((wm >> 4) + t) * KT * 1024 + lane * 16;
        pB[t] = to_global(Wq) + (size_t)((wn >> 4) + t) * KT * 1024 + lane * 16;
    }

    v8i acc[4][4];
    #pragma unroll
    for (int mt = 0; mt < 4; ++mt)
        #pragma unroll
        for (int nt = 0; nt < 4; ++nt)
            #pragma unroll
            for (int r = 0; r < 8; ++r)
                acc[mt][nt][r] = 0;

    for (int kt = 0; kt < KT; ++kt) {
        const int kb = kt << 10;
        v8i a[4], bf[4];
        #pragma unroll
        for (int mt = 0; mt < 4; ++mt) {
            v4i lo = gload16(pA[mt] + kb);
            v4i hi = gload16(pA[mt] + kb + 512);
            v8i t;
            t[0] = lo[0]; t[1] = lo[1]; t[2] = lo[2]; t[3] = lo[3];
            t[4] = hi[0]; t[5] = hi[1]; t[6] = hi[2]; t[7] = hi[3];
            a[mt] = t;
        }
        #pragma unroll
        for (int nt = 0; nt < 4; ++nt) {
            v4i lo = gload16(pB[nt] + kb);
            v4i hi = gload16(pB[nt] + kb + 512);
            v8i t;
            t[0] = lo[0]; t[1] = lo[1]; t[2] = lo[2]; t[3] = lo[3];
            t[4] = hi[0]; t[5] = hi[1]; t[6] = hi[2]; t[7] = hi[3];
            bf[nt] = t;
        }
        #pragma unroll
        for (int mt = 0; mt < 4; ++mt)
            #pragma unroll
            for (int nt = 0; nt < 4; ++nt)
                acc[mt][nt] = __builtin_amdgcn_wmma_i32_16x16x64_iu8(
                     true, a[mt],  true, bf[nt],
                    acc[mt][nt],  false,  false);
    }

    #pragma unroll
    for (int mt = 0; mt < 4; ++mt) {
        float sa[8];
        #pragma unroll
        for (int r = 0; r < 8; ++r)
            sa[r] = sact[wm + mt * 16 + lhi * 8 + r];
        #pragma unroll
        for (int nt = 0; nt < 4; ++nt) {
            const int n    = wn + nt * 16 + l15;
            const float ws = wscale[n];
            const float bb = bias[n];
            #pragma unroll
            for (int r = 0; r < 8; ++r) {
                const int m = wm + mt * 16 + lhi * 8 + r;
                out[(size_t)m * N + n] = (float)acc[mt][nt][r] * sa[r] * ws + bb;
            }
        }
    }
}

extern "C" void kernel_launch(void* const* d_in, const int* in_sizes, int n_in,
                              void* d_out, int out_size, void* d_ws, size_t ws_size,
                              hipStream_t stream) {
    const float* x       = (const float*)d_in[0];
    const int*   w_q     = (const int*)d_in[1];
    const float* w_scale = (const float*)d_in[2];
    const float* b       = (const float*)d_in[3];
    float*       out     = (float*)d_out;

    const int N = in_sizes[2];
    const int K = in_sizes[1] / N;
    const int M = in_sizes[0] / K;

    signed char* Aq  = (signed char*)d_ws;
    signed char* Wq8 = Aq + (size_t)M * K;
    float*       sct = (float*)(Wq8 + (size_t)N * K);

    quantize_act<<<M, 256, 0, stream>>>(x, Aq, sct, K);

    dim3 pgrid((K / 8 + 255) / 256, N);
    pack_w<<<pgrid, 256, 0, stream>>>(w_q, Wq8, K);

    dim3 grid(N / 256, M / 128);
    w8a8_gemm<<<grid, 256, 0, stream>>>(Aq, Wq8, sct, w_scale, b, out, M, N, K);
}
